// GCNEncoder_22110491639904
// MI455X (gfx1250) — compile-verified, matrix operand order checked by reading, not run on hardware
//
#include <hip/hip_runtime.h>
#include <math.h>

typedef __attribute__((ext_vector_type(16))) _Float16 v16h;
typedef __attribute__((ext_vector_type(8)))  float    v8f;

#define NNODES 100000
#define NEDGES 1600000
#define FDIM   128

static inline int ceil_div(long long a, long long b) { return (int)((a + b - 1) / b); }

// ---------------------------------------------------------------------------
// Degree / norm kernels
// ---------------------------------------------------------------------------
__global__ void init_deg_kernel(float* __restrict__ odeg, float* __restrict__ ideg, int n) {
  int i = blockIdx.x * blockDim.x + threadIdx.x;
  if (i < n) { odeg[i] = 1.0f; ideg[i] = 1.0f; }   // self-loop contributes 1 to both
}

__global__ void degree_kernel(const int* __restrict__ src, const int* __restrict__ dst,
                              float* __restrict__ odeg, float* __restrict__ ideg, int e) {
  int i = blockIdx.x * blockDim.x + threadIdx.x;
  if (i < e) {
    atomicAdd(&odeg[src[i]], 1.0f);
    atomicAdd(&ideg[dst[i]], 1.0f);
  }
}

__global__ void rsqrt_kernel(float* __restrict__ a, float* __restrict__ b, int n) {
  int i = blockIdx.x * blockDim.x + threadIdx.x;
  if (i < n) {
    a[i] = rsqrtf(fmaxf(a[i], 1.0f));
    b[i] = rsqrtf(fmaxf(b[i], 1.0f));
  }
}

__global__ void zero_kernel(float* __restrict__ p, int n) {
  int i = blockIdx.x * blockDim.x + threadIdx.x;
  if (i < n) p[i] = 0.0f;
}

// ---------------------------------------------------------------------------
// Weight pre-swizzle: W [128 x nout] f32 row-major  ->  f16 WMMA B-fragments.
// Fragment layout (16-bit B, 32x16, wave32): lane l holds column (l&15);
// element j holds K = kc*32 + (l>>4)*8 + (j&7) + 16*(j>>3).
// Stored so lane l of (kchunk kc, ntile t) reads 16 contiguous halfs.
// ---------------------------------------------------------------------------
__global__ void pack_w_kernel(const float* __restrict__ W, _Float16* __restrict__ Wp, int nout) {
  int idx = blockIdx.x * blockDim.x + threadIdx.x;
  int nt = nout >> 4;
  int total = 4 * nt * 32 * 16;            // 4 K-chunks of 32
  if (idx >= total) return;
  int j    = idx & 15;
  int l    = (idx >> 4) & 31;
  int rest = idx >> 9;
  int t    = rest % nt;
  int c    = rest / nt;
  int col   = t * 16 + (l & 15);
  int kbase = c * 32 + ((l >> 4) * 8);
  int K     = kbase + (j & 7) + ((j >> 3) * 16);
  Wp[idx] = (_Float16)W[K * nout + col];
}

// ---------------------------------------------------------------------------
// WMMA GEMM:  H[N x NOUT] = (X * onorm[:,None]) @ W   (f16 inputs, f32 accum)
// One wave per 16x16 tile; K = 128 = 4 chunks of 32.
// ---------------------------------------------------------------------------
template<int NOUT>
__global__ void gemm_wmma_kernel(const float* __restrict__ X,
                                 const float* __restrict__ onorm,
                                 const _Float16* __restrict__ Wp,
                                 float* __restrict__ H) {
  constexpr int NT = NOUT / 16;
  int wave = (int)((blockIdx.x * blockDim.x + threadIdx.x) >> 5);
  int lane = threadIdx.x & 31;
  int mtile = wave / NT;
  int ntile = wave % NT;
  if (mtile * 16 >= NNODES) return;        // wave-uniform; never hit (100000 % 16 == 0)

  int r    = lane & 15;                    // row within tile (A) / col within tile (C)
  int half = lane >> 4;
  int row  = mtile * 16 + r;
  float scale = onorm[row];
  const float* px = X + (size_t)row * FDIM;

  v8f acc = {0.f, 0.f, 0.f, 0.f, 0.f, 0.f, 0.f, 0.f};

#pragma unroll
  for (int kc = 0; kc < 4; ++kc) {
    int kbase = kc * 32 + half * 8;
    float4 f0 = *(const float4*)(px + kbase);
    float4 f1 = *(const float4*)(px + kbase + 4);
    float4 f2 = *(const float4*)(px + kbase + 16);
    float4 f3 = *(const float4*)(px + kbase + 20);
    v16h a;
    a[0]  = (_Float16)(f0.x * scale); a[1]  = (_Float16)(f0.y * scale);
    a[2]  = (_Float16)(f0.z * scale); a[3]  = (_Float16)(f0.w * scale);
    a[4]  = (_Float16)(f1.x * scale); a[5]  = (_Float16)(f1.y * scale);
    a[6]  = (_Float16)(f1.z * scale); a[7]  = (_Float16)(f1.w * scale);
    a[8]  = (_Float16)(f2.x * scale); a[9]  = (_Float16)(f2.y * scale);
    a[10] = (_Float16)(f2.z * scale); a[11] = (_Float16)(f2.w * scale);
    a[12] = (_Float16)(f3.x * scale); a[13] = (_Float16)(f3.y * scale);
    a[14] = (_Float16)(f3.z * scale); a[15] = (_Float16)(f3.w * scale);

    v16h b = *(const v16h*)(Wp + ((size_t)(kc * NT + ntile) * 32 + lane) * 16);

    acc = __builtin_amdgcn_wmma_f32_16x16x32_f16(
        /*neg_a=*/false, a, /*neg_b=*/false, b,
        /*c_mod=*/(short)0, acc, /*reuse_a=*/false, /*reuse_b=*/false);
  }

  // C/D layout: lanes 0-15 -> M=r_vgpr, lanes 16-31 -> M=8+r_vgpr; N = lane&15
  int col = ntile * 16 + r;
  float* ph = H + (size_t)(mtile * 16 + half * 8) * NOUT + col;
#pragma unroll
  for (int vr = 0; vr < 8; ++vr)
    ph[(size_t)vr * NOUT] = acc[vr];
}

// ---------------------------------------------------------------------------
// Edge aggregation: one wave per edge; lane owns NOUT/32 contiguous floats.
// agg[dst] += h[src]   (L2-resident gather + f32 atomics)
// ---------------------------------------------------------------------------
template<int NOUT>
__global__ void edge_agg_kernel(const int* __restrict__ src, const int* __restrict__ dst,
                                const float* __restrict__ h, float* __restrict__ agg, int e) {
  int wid  = (int)((blockIdx.x * blockDim.x + threadIdx.x) >> 5);
  int lane = threadIdx.x & 31;
  if (wid >= e) return;
  int s = src[wid];
  int d = dst[wid];
  constexpr int V = NOUT / 32;             // 4 (dout=128) or 2 (dout=64)
  const float* hp = h   + (size_t)s * NOUT + lane * V;
  float*       ap = agg + (size_t)d * NOUT + lane * V;
  float v[V];
#pragma unroll
  for (int k = 0; k < V; ++k) v[k] = hp[k];
#pragma unroll
  for (int k = 0; k < V; ++k) atomicAdd(ap + k, v[k]);
}

// ---------------------------------------------------------------------------
// Finalize: out = (agg + h_selfloop) * in_norm + b  (+ReLU)
// ---------------------------------------------------------------------------
template<int NOUT, bool RELU>
__global__ void finalize_kernel(const float* __restrict__ agg, const float* __restrict__ h,
                                const float* __restrict__ inorm, const float* __restrict__ b,
                                float* __restrict__ out, int n) {
  int idx = blockIdx.x * blockDim.x + threadIdx.x;
  if (idx >= n) return;
  int node = idx / NOUT;
  int dcol = idx % NOUT;
  float v = (agg[idx] + h[idx]) * inorm[node] + b[dcol];
  out[idx] = RELU ? fmaxf(v, 0.0f) : v;
}

// ---------------------------------------------------------------------------
// Orchestration
// ---------------------------------------------------------------------------
extern "C" void kernel_launch(void* const* d_in, const int* in_sizes, int n_in,
                              void* d_out, int out_size, void* d_ws, size_t ws_size,
                              hipStream_t stream) {
  const float* feat = (const float*)d_in[0];
  const int*   src  = (const int*)  d_in[1];
  const int*   dst  = (const int*)  d_in[2];
  const float* W1   = (const float*)d_in[3];
  const float* b1   = (const float*)d_in[4];
  const float* W2   = (const float*)d_in[5];
  const float* b2   = (const float*)d_in[6];
  const float* W3   = (const float*)d_in[7];
  const float* b3   = (const float*)d_in[8];
  float* out = (float*)d_out;

  // Workspace carve-out (256B aligned slabs)
  size_t off = 0;
  auto alloc = [&](size_t bytes) -> char* {
    char* p = (char*)d_ws + off;
    off += (bytes + 255) & ~(size_t)255;
    return p;
  };
  float*     onorm = (float*)alloc((size_t)NNODES * sizeof(float));
  float*     inorm = (float*)alloc((size_t)NNODES * sizeof(float));
  float*     hbuf  = (float*)alloc((size_t)NNODES * FDIM * sizeof(float));
  float*     abuf  = (float*)alloc((size_t)NNODES * FDIM * sizeof(float));
  float*     xbuf  = (float*)alloc((size_t)NNODES * FDIM * sizeof(float));
  _Float16*  Wp    = (_Float16*)alloc((size_t)FDIM * FDIM * sizeof(_Float16));

  const int TB = 256;

  // ---- norms ----
  init_deg_kernel<<<ceil_div(NNODES, TB), TB, 0, stream>>>(onorm, inorm, NNODES);
  degree_kernel<<<ceil_div(NEDGES, TB), TB, 0, stream>>>(src, dst, onorm, inorm, NEDGES);
  rsqrt_kernel<<<ceil_div(NNODES, TB), TB, 0, stream>>>(onorm, inorm, NNODES);

  // ---- layer 1: feat -> xbuf (relu) ----
  pack_w_kernel<<<ceil_div(4 * 8 * 512, TB), TB, 0, stream>>>(W1, Wp, 128);
  zero_kernel<<<ceil_div((long long)NNODES * 128, TB), TB, 0, stream>>>(abuf, NNODES * 128);
  {
    int waves = (NNODES / 16) * (128 / 16);
    gemm_wmma_kernel<128><<<ceil_div((long long)waves * 32, TB), TB, 0, stream>>>(feat, onorm, Wp, hbuf);
  }
  edge_agg_kernel<128><<<ceil_div((long long)NEDGES * 32, TB), TB, 0, stream>>>(src, dst, hbuf, abuf, NEDGES);
  finalize_kernel<128, true><<<ceil_div((long long)NNODES * 128, TB), TB, 0, stream>>>(
      abuf, hbuf, inorm, b1, xbuf, NNODES * 128);

  // ---- layer 2: xbuf -> xbuf (relu) ----
  pack_w_kernel<<<ceil_div(4 * 8 * 512, TB), TB, 0, stream>>>(W2, Wp, 128);
  zero_kernel<<<ceil_div((long long)NNODES * 128, TB), TB, 0, stream>>>(abuf, NNODES * 128);
  {
    int waves = (NNODES / 16) * (128 / 16);
    gemm_wmma_kernel<128><<<ceil_div((long long)waves * 32, TB), TB, 0, stream>>>(xbuf, onorm, Wp, hbuf);
  }
  edge_agg_kernel<128><<<ceil_div((long long)NEDGES * 32, TB), TB, 0, stream>>>(src, dst, hbuf, abuf, NEDGES);
  finalize_kernel<128, true><<<ceil_div((long long)NNODES * 128, TB), TB, 0, stream>>>(
      abuf, hbuf, inorm, b2, xbuf, NNODES * 128);

  // ---- layer 3: xbuf -> d_out (no relu), dout = 64 ----
  pack_w_kernel<<<ceil_div(4 * 4 * 512, TB), TB, 0, stream>>>(W3, Wp, 64);
  zero_kernel<<<ceil_div((long long)NNODES * 64, TB), TB, 0, stream>>>(abuf, NNODES * 64);
  {
    int waves = (NNODES / 16) * (64 / 16);
    gemm_wmma_kernel<64><<<ceil_div((long long)waves * 32, TB), TB, 0, stream>>>(xbuf, onorm, Wp, hbuf);
  }
  edge_agg_kernel<64><<<ceil_div((long long)NEDGES * 32, TB), TB, 0, stream>>>(src, dst, hbuf, abuf, NEDGES);
  finalize_kernel<64, false><<<ceil_div((long long)NNODES * 64, TB), TB, 0, stream>>>(
      abuf, hbuf, inorm, b3, out, NNODES * 64);
}
